// HNet_23467701305549
// MI455X (gfx1250) — matrix operand order repaired and checked by reading, not compiled, not run on hardware
//
#include <hip/hip_runtime.h>
#include <hip/hip_bf16.h>

// ---------------------------------------------------------------------------
// H-Net forward on MI455X (gfx1250, wave32).
// 6 bf16 WMMA GEMMs (enc, 4 fused projections, decoder) + router + chunked
// linear-recurrence scan. Matrix math via v_wmma_f32_16x16x32_bf16; A tiles
// staged into double-buffered LDS with global_load_async_to_lds_b128
// (ASYNCcnt) so fills overlap WMMA compute.
// ---------------------------------------------------------------------------

#define D_DIM   1024
#define NSEQ    4
#define SEQ_LEN 8192
#define L_TOK   (NSEQ * SEQ_LEN)
#define REPS    1e-4f
#define CH      512
#define NCH     (L_TOK / CH)

typedef __attribute__((ext_vector_type(16))) __bf16 v16bf;
typedef __attribute__((ext_vector_type(8)))  __bf16 v8bf;
typedef __attribute__((ext_vector_type(8)))  float  v8f;

// ---------------------------------------------------------------------------
// Weight conversion f32 -> bf16 (weights stay hot in 192MB L2 afterwards)
// ---------------------------------------------------------------------------
__global__ __launch_bounds__(256) void cvt_w_k(
    const float* __restrict__ W0, const float* __restrict__ W1,
    const float* __restrict__ W2, const float* __restrict__ W3,
    const float* __restrict__ W4, const float* __restrict__ W5,
    __bf16* O0, __bf16* O1, __bf16* O2, __bf16* O3, __bf16* O4, __bf16* O5)
{
    const float* srcs[6] = {W0, W1, W2, W3, W4, W5};
    __bf16*      dsts[6] = {O0, O1, O2, O3, O4, O5};
    int w = blockIdx.y;
    size_t i = (size_t)blockIdx.x * 256 + threadIdx.x;
    dsts[w][i] = (__bf16)srcs[w][i];
}

// ---------------------------------------------------------------------------
// Tiled WMMA GEMM:  O[p] = A @ B[p]^T   (B[p] is [N,K] row-major, = weight)
// Block: 256 threads = 8 waves arranged 4(M) x 2(N).
// Wave tile: (16*WM) x (16*WN).  Block tile: (64*WM) x (32*WN).
// A staged in double-buffered LDS (padded stride 40 vs 64 banks). For bf16 A
// the fill uses global_load_async_to_lds_b128; the f32-A variant converts
// through VGPRs (v_cvt_pk_bf16_f32 + ds_store). B fragments read directly
// from global: each weight is 2MB bf16, resident in the 192MB L2.
// NPROJ>1 fuses multiple weight matrices over one A-tile load.
// ---------------------------------------------------------------------------
template<int NPROJ, int WM, int WN, bool OUTF32, bool AF32>
__global__ __launch_bounds__(256) void wmma_gemm_k(
    const void* __restrict__ Ap,
    const __bf16* __restrict__ B0, const __bf16* __restrict__ B1,
    const __bf16* __restrict__ B2, const __bf16* __restrict__ B3,
    void* __restrict__ O0, void* __restrict__ O1,
    void* __restrict__ O2, void* __restrict__ O3,
    int M, int N, int K)
{
    constexpr int BM = 64 * WM;
    constexpr int BN = 32 * WN;
    __shared__ __bf16 ldsA[2][BM][40];   // 32 cols + 8 pad, double buffered

    const int tid   = threadIdx.x;
    const int lane  = tid & 31;
    const int wave  = tid >> 5;
    const int waveM = wave & 3;       // 0..3
    const int waveN = wave >> 2;      // 0..1
    const int m0    = blockIdx.x * BM;
    const int n0    = blockIdx.y * BN;

    const float*  Af = (const float*)Ap;
    const __bf16* Ab = (const __bf16*)Ap;
    const __bf16* Bs[4] = {B0, B1, B2, B3};

    v8f acc[NPROJ][WM][WN];
#pragma unroll
    for (int p = 0; p < NPROJ; ++p)
#pragma unroll
        for (int i = 0; i < WM; ++i)
#pragma unroll
            for (int j = 0; j < WN; ++j)
#pragma unroll
                for (int e = 0; e < 8; ++e) acc[p][i][j][e] = 0.0f;

    // Stage one BMx32 bf16 tile. Tile = BM*64 bytes; 256 threads handle one
    // 16B chunk each, CPT = BM/64 chunks per thread (1 or 2). No divergence.
    auto fillTile = [&](int buf, int k0) {
        constexpr int CPT = BM / 64;
#pragma unroll
        for (int cc = 0; cc < CPT; ++cc) {
            int chunk = tid + cc * 256;
            int r  = chunk >> 2;          // row in tile
            int c8 = (chunk & 3) * 8;     // col (bf16 elems), 8 per 16B chunk
            size_t goff = (size_t)(m0 + r) * K + k0 + c8;
            if constexpr (AF32) {
                const float* src = Af + goff;
                v8bf tmp;
#pragma unroll
                for (int i = 0; i < 8; ++i) tmp[i] = (__bf16)src[i];
                *(v8bf*)&ldsA[buf][r][c8] = tmp;
            } else {
                // async memory -> LDS, tracked by ASYNCcnt (no VGPR data path)
                unsigned int ldsoff =
                    (unsigned int)(uintptr_t)&ldsA[buf][r][c8];
                unsigned long long ga =
                    (unsigned long long)(uintptr_t)(Ab + goff);
                asm volatile("global_load_async_to_lds_b128 %0, %1, off"
                             :: "v"(ldsoff), "v"(ga) : "memory");
            }
        }
    };

    // A-fragment layout (16-bit 16x32): lane = row M, K contiguous kb..kb+15,
    // kb = (lane>>4)*16.  B-fragment (32x16): lane n = lane&15,
    // K contiguous [kb2, kb2+16), kb2 = (lane>>4)*16  -> one 32B row chunk of W.
    const int ln  = lane & 15;
    const int kb  = (lane >> 4) * 16;
    const int kb2 = (lane >> 4) * 16;

    // Prologue: fill buffer 0, make it visible to the whole block.
    fillTile(0, 0);
    if constexpr (!AF32)
        asm volatile("s_wait_asynccnt 0x0" ::: "memory");
    __syncthreads();

    for (int ki = 0; ki < K / 32; ++ki) {
        const int k0  = ki * 32;
        const int buf = ki & 1;

        // Kick off the next tile's fill before computing this one.
        if (k0 + 32 < K) fillTile(buf ^ 1, k0 + 32);

        // ---- A fragments from LDS ----
        v16bf afrag[WM];
#pragma unroll
        for (int wm = 0; wm < WM; ++wm) {
            int mr = waveM * 16 * WM + wm * 16 + ln;
            v8bf lo = *(const v8bf*)&ldsA[buf][mr][kb];
            v8bf hi = *(const v8bf*)&ldsA[buf][mr][kb + 8];
#pragma unroll
            for (int i = 0; i < 8; ++i) { afrag[wm][i] = lo[i]; afrag[wm][8 + i] = hi[i]; }
        }

        // ---- B fragments (global, L2-hot) + WMMA ----
#pragma unroll
        for (int p = 0; p < NPROJ; ++p) {
#pragma unroll
            for (int wn = 0; wn < WN; ++wn) {
                int ng = n0 + waveN * 16 * WN + wn * 16 + ln;
                v16bf bfrag = *(const v16bf*)&Bs[p][(size_t)ng * K + k0 + kb2];
#pragma unroll
                for (int wm = 0; wm < WM; ++wm) {
                    acc[p][wm][wn] = __builtin_amdgcn_wmma_f32_16x16x32_bf16(
                        false, afrag[wm], false, bfrag,
                        (short)0, acc[p][wm][wn], false, false);
                }
            }
        }

        // Next buffer must be fully landed (own async ops) and every wave done
        // with this buffer (their ds reads complete before their wmma waits).
        if (k0 + 32 < K) {
            if constexpr (!AF32)
                asm volatile("s_wait_asynccnt 0x0" ::: "memory");
            __syncthreads();
        }
    }

    // ---- store C/D (lane n = lane&15, VGPR j -> row j + 8*(lane>>4)) ----
    void* Os[4] = {O0, O1, O2, O3};
#pragma unroll
    for (int p = 0; p < NPROJ; ++p) {
#pragma unroll
        for (int wm = 0; wm < WM; ++wm) {
#pragma unroll
            for (int wn = 0; wn < WN; ++wn) {
                int col = n0 + waveN * 16 * WN + wn * 16 + ln;
#pragma unroll
                for (int j = 0; j < 8; ++j) {
                    size_t row  = (size_t)m0 + waveM * 16 * WM + wm * 16 + j + 8 * (lane >> 4);
                    size_t oidx = row * (size_t)N + col;
                    float v = acc[p][wm][wn][j];
                    if constexpr (OUTF32) ((float*)Os[p])[oidx]  = v;
                    else                  ((__bf16*)Os[p])[oidx] = (__bf16)v;
                }
            }
        }
    }
}

// ---------------------------------------------------------------------------
// Router: one wave per token. p[l] = 0.5*(1 - cos(q[l-1], k[l])), forced to 1
// at sequence starts; emits a[l] and bscale[l] for the recurrence.
// ---------------------------------------------------------------------------
__global__ __launch_bounds__(256) void router_k(
    const __bf16* __restrict__ qB, const __bf16* __restrict__ kB,
    const int* __restrict__ cs,
    float* __restrict__ aArr, float* __restrict__ bsArr)
{
    int l    = blockIdx.x * 8 + (threadIdx.x >> 5);
    int lane = threadIdx.x & 31;

    bool start = (l == 0);
#pragma unroll
    for (int s = 0; s < NSEQ; ++s) start |= (l == cs[s]);

    float p;
    if (start) {
        p = 1.0f;
    } else {
        float dq = 0.f, nq = 0.f, nk = 0.f;
        const __bf16* qrow = qB + (size_t)(l - 1) * D_DIM;
        const __bf16* krow = kB + (size_t)l * D_DIM;
#pragma unroll
        for (int j = 0; j < 4; ++j) {
            v8bf qv = *(const v8bf*)&qrow[j * 256 + lane * 8];
            v8bf kv = *(const v8bf*)&krow[j * 256 + lane * 8];
#pragma unroll
            for (int i = 0; i < 8; ++i) {
                float qf = (float)qv[i], kf = (float)kv[i];
                dq = fmaf(qf, kf, dq);
                nq = fmaf(qf, qf, nq);
                nk = fmaf(kf, kf, nk);
            }
        }
#pragma unroll
        for (int off = 16; off > 0; off >>= 1) {
            dq += __shfl_xor(dq, off, 32);
            nq += __shfl_xor(nq, off, 32);
            nk += __shfl_xor(nk, off, 32);
        }
        float qn = fmaxf(sqrtf(nq), 1e-12f);
        float kn = fmaxf(sqrtf(nk), 1e-12f);
        p = 0.5f * (1.0f - dq / (qn * kn));
    }

    if (lane == 0) {
        bool  mask = p > 0.5f;
        float pc   = fminf(fmaxf(p, REPS), 1.0f - REPS);
        float a    = start ? 0.0f : (mask ? 1.0f - pc : 1.0f);
        float bs   = mask ? pc : 0.0f;
        aArr[l]  = a;
        bsArr[l] = bs;
    }
}

// ---------------------------------------------------------------------------
// Chunked scan for h_t = a_t*h_{t-1} + bs_t*m_t   (a scalar, m in R^D).
// a=0 at sequence starts handles resets for free.
// ---------------------------------------------------------------------------
__global__ __launch_bounds__(256) void chunk_prod_k(
    const float* __restrict__ aArr, float* __restrict__ prodA)
{
    int c    = blockIdx.x * 8 + (threadIdx.x >> 5);
    int lane = threadIdx.x & 31;
    float pr = 1.0f;
    int t0 = c * CH + lane * 16;
#pragma unroll
    for (int j = 0; j < 16; ++j) pr *= aArr[t0 + j];
#pragma unroll
    for (int off = 16; off > 0; off >>= 1) pr *= __shfl_xor(pr, off, 32);
    if (lane == 0) prodA[c] = pr;
}

__global__ __launch_bounds__(256) void chunk_scan_k(
    const float* __restrict__ aArr, const float* __restrict__ bsArr,
    const __bf16* __restrict__ mB, float* __restrict__ Bchunk)
{
    int d = blockIdx.x * 256 + threadIdx.x;
    int c = blockIdx.y;
    float h = 0.0f;
    int t0 = c * CH;
    for (int t = t0; t < t0 + CH; ++t) {
        float mv = (float)mB[(size_t)t * D_DIM + d];
        h = fmaf(aArr[t], h, bsArr[t] * mv);
    }
    Bchunk[(size_t)c * D_DIM + d] = h;
}

__global__ __launch_bounds__(256) void carry_k(
    const float* __restrict__ prodA, const float* __restrict__ Bchunk,
    float* __restrict__ carryIn)
{
    int d = blockIdx.x * 256 + threadIdx.x;
    float H = 0.0f;
    for (int c = 0; c < NCH; ++c) {
        carryIn[(size_t)c * D_DIM + d] = H;
        H = prodA[c] * H + Bchunk[(size_t)c * D_DIM + d];
    }
}

__global__ __launch_bounds__(256) void apply_k(
    const float* __restrict__ aArr, const float* __restrict__ bsArr,
    const __bf16* __restrict__ mB, const float* __restrict__ carryIn,
    const __bf16* __restrict__ resB, __bf16* __restrict__ ycombB)
{
    int d = blockIdx.x * 256 + threadIdx.x;
    int c = blockIdx.y;
    float h = carryIn[(size_t)c * D_DIM + d];
    int t0 = c * CH;
    for (int t = t0; t < t0 + CH; ++t) {
        size_t idx = (size_t)t * D_DIM + d;
        h = fmaf(aArr[t], h, bsArr[t] * (float)mB[idx]);
        ycombB[idx] = (__bf16)(h + (float)resB[idx]);  // fuse + residual
    }
}

// ---------------------------------------------------------------------------
extern "C" void kernel_launch(void* const* d_in, const int* in_sizes, int n_in,
                              void* d_out, int out_size, void* d_ws, size_t ws_size,
                              hipStream_t stream)
{
    (void)in_sizes; (void)n_in; (void)out_size; (void)ws_size;

    const float* x  = (const float*)d_in[0];
    const float* Wq = (const float*)d_in[1];
    const float* Wk = (const float*)d_in[2];
    const float* Wr = (const float*)d_in[3];
    const float* We = (const float*)d_in[4];
    const float* Wm = (const float*)d_in[5];
    const float* Wd = (const float*)d_in[6];
    const int*   cs = (const int*)d_in[7];
    float* out = (float*)d_out;

    char* ws = (char*)d_ws;
    size_t off = 0;
    auto alloc = [&](size_t bytes) -> char* {
        char* p = ws + off;
        off = (off + bytes + 255) & ~(size_t)255;
        return p;
    };

    const size_t WB = (size_t)D_DIM * D_DIM * sizeof(__bf16);
    const size_t LB = (size_t)L_TOK * D_DIM * sizeof(__bf16);
    __bf16* WqB  = (__bf16*)alloc(WB);
    __bf16* WkB  = (__bf16*)alloc(WB);
    __bf16* WrB  = (__bf16*)alloc(WB);
    __bf16* WeB  = (__bf16*)alloc(WB);
    __bf16* WmB  = (__bf16*)alloc(WB);
    __bf16* WdB  = (__bf16*)alloc(WB);
    __bf16* encB = (__bf16*)alloc(LB);
    __bf16* qB   = (__bf16*)alloc(LB);
    __bf16* kB   = (__bf16*)alloc(LB);
    __bf16* mB   = (__bf16*)alloc(LB);
    __bf16* resB = (__bf16*)alloc(LB);
    float* aArr  = (float*)alloc((size_t)L_TOK * 4);
    float* bsArr = (float*)alloc((size_t)L_TOK * 4);
    float* prodA = (float*)alloc((size_t)NCH * 4);
    float* Bch   = (float*)alloc((size_t)NCH * D_DIM * 4);
    float* carry = (float*)alloc((size_t)NCH * D_DIM * 4);
    __bf16* ycombB = qB;   // q/k dead after router; reuse for (y + residual)

    // 1) weights -> bf16
    cvt_w_k<<<dim3(D_DIM * D_DIM / 256, 6), 256, 0, stream>>>(
        Wq, Wk, Wr, We, Wm, Wd, WqB, WkB, WrB, WeB, WmB, WdB);

    // 2) enc = x @ We^T   (f32 A converted while staging to LDS)
    wmma_gemm_k<1, 2, 2, false, true>
        <<<dim3(L_TOK / 128, D_DIM / 64), 256, 0, stream>>>(
            x, WeB, WeB, WeB, WeB, encB, encB, encB, encB,
            L_TOK, D_DIM, D_DIM);

    // 3) fused 4 projections of enc: residual/q/k/m  (A tile loaded once,
    //    async into LDS)
    wmma_gemm_k<4, 1, 1, false, false>
        <<<dim3(L_TOK / 64, D_DIM / 32), 256, 0, stream>>>(
            encB, WrB, WqB, WkB, WmB, resB, qB, kB, mB,
            L_TOK, D_DIM, D_DIM);

    // 4) router -> a[l], bscale[l]
    router_k<<<L_TOK / 8, 256, 0, stream>>>(qB, kB, cs, aArr, bsArr);

    // 5) chunked associative scan
    chunk_prod_k<<<NCH / 8, 256, 0, stream>>>(aArr, prodA);
    chunk_scan_k<<<dim3(D_DIM / 256, NCH), 256, 0, stream>>>(aArr, bsArr, mB, Bch);
    carry_k<<<D_DIM / 256, 256, 0, stream>>>(prodA, Bch, carry);
    apply_k<<<dim3(D_DIM / 256, NCH), 256, 0, stream>>>(
        aArr, bsArr, mB, carry, resB, ycombB);

    // 6) out = (y + residual) @ Wd^T  -> f32 (async A staging)
    wmma_gemm_k<1, 2, 2, true, false>
        <<<dim3(L_TOK / 128, D_DIM / 64), 256, 0, stream>>>(
            ycombB, WdB, WdB, WdB, WdB, out, out, out, out,
            L_TOK, D_DIM, D_DIM);
}
